// MaskedSelfAttention_35983236006061
// MI455X (gfx1250) — matrix operand order repaired and checked by reading, not compiled, not run on hardware
//
#include <hip/hip_runtime.h>

// ---------------------------------------------------------------------------
// MaskedSelfAttention on MI455X (gfx1250): all GEMMs via v_wmma_f32_16x16x32_bf16
// B=4, L=1024, D=1024, H=16, DH=64
// ---------------------------------------------------------------------------

typedef __attribute__((ext_vector_type(16))) __bf16 v16bf;
typedef __attribute__((ext_vector_type(8)))  float  v8f;

union Frag {
    v16bf v;
    unsigned u[8];
};

__device__ __forceinline__ unsigned short f2bf(float f) {
    unsigned u = __float_as_uint(f);
    return (unsigned short)((u + 0x7FFFu + ((u >> 16) & 1u)) >> 16);  // RNE
}

// ---------------------------------------------------------------------------
// Core tile: 256 threads = 8 wave32s arranged 4 (M) x 2 (N).
// NT = number of 16-wide accumulators per wave:
//   NT=2 -> workgroup tile 64x64,  2 WMMA per A-frag (used when N==64)
//   NT=4 -> workgroup tile 64x128, 4 WMMA per A-frag (main GEMMs)
// A: bf16 [M,K] row-major (or f32, converted during staging).
// B: bf16 [K,N] row-major, staged transposed into LDS as [n][k].
// ---------------------------------------------------------------------------
template<int NT, bool AF32>
__device__ __forceinline__ void wmma_tile(
    const void* __restrict__ Ap, int ldA,
    const unsigned short* __restrict__ Bp, int ldB,
    int Kdim, int rowBase, int colBase, v8f (&c)[NT])
{
    constexpr int W = NT * 32;                 // workgroup N width (64 or 128)
    __shared__ unsigned short lA[64 * 32];
    __shared__ unsigned short lB[W * 32];

    const int tid  = threadIdx.x;
    const int lane = tid & 31;
    const int wave = tid >> 5;
    const int wm   = (wave >> 1) * 16;         // wave M offset within 64
    const int wn   = (wave & 1) * (NT * 16);   // wave N offset within W
    const int l15  = lane & 15;
    const int hi   = lane >> 4;

#pragma unroll
    for (int t = 0; t < NT; ++t)
#pragma unroll
        for (int i = 0; i < 8; ++i) c[t][i] = 0.0f;

    for (int kb = 0; kb < Kdim; kb += 32) {
        // ---- stage A tile (64 rows x 32 k) into lA, row-major ----
        {
            const int r  = tid >> 2;
            const int c8 = (tid & 3) * 8;
            if (AF32) {
                const float* ga = (const float*)Ap + (size_t)(rowBase + r) * ldA + kb + c8;
                float4 f0 = *(const float4*)ga;
                float4 f1 = *(const float4*)(ga + 4);
                unsigned short* d = &lA[r * 32 + c8];
                d[0] = f2bf(f0.x); d[1] = f2bf(f0.y); d[2] = f2bf(f0.z); d[3] = f2bf(f0.w);
                d[4] = f2bf(f1.x); d[5] = f2bf(f1.y); d[6] = f2bf(f1.z); d[7] = f2bf(f1.w);
                if (kb + 32 < Kdim) __builtin_prefetch((const char*)(ga + ldA ? ga + 32 : ga), 0, 1);
            } else {
                const unsigned short* ga =
                    (const unsigned short*)Ap + (size_t)(rowBase + r) * ldA + kb + c8;
                *(uint4*)&lA[r * 32 + c8] = *(const uint4*)ga;
                if (kb + 32 < Kdim) __builtin_prefetch((const char*)(ga + 32), 0, 1);
            }
        }
        // ---- stage B tile (32 k x W n), transposed into lB[n][k] ----
#pragma unroll
        for (int ch = 0; ch < W / 64; ++ch) {
            const int f   = (ch * 256 + tid) * 8;
            const int kr  = f / W;
            const int col = f & (W - 1);
            const unsigned short* gb = Bp + (size_t)(kb + kr) * ldB + colBase + col;
            unsigned short tmp[8];
            *(uint4*)tmp = *(const uint4*)gb;
#pragma unroll
            for (int j = 0; j < 8; ++j) lB[(col + j) * 32 + kr] = tmp[j];
            if (kb + 32 < Kdim) __builtin_prefetch((const char*)(gb + (size_t)32 * ldB), 0, 1);
        }
        __syncthreads();

        // ---- build fragments per ISA 16-bit wave32 layouts ----
        Frag a;
#pragma unroll
        for (int p = 0; p < 8; ++p) {   // A: VGPR p <- K pair {g*16 + hi*8 + 2j, +1}
            const int kk = (p >> 2) * 16 + hi * 8 + (p & 3) * 2;
            a.u[p] = *(const unsigned*)&lA[(wm + l15) * 32 + kk];
        }
        Frag b[NT];
#pragma unroll
        for (int t = 0; t < NT; ++t)
#pragma unroll
            for (int p = 0; p < 8; ++p) {  // B: VGPR p <- K pair {g*16 + hi*8 + 2j, +1}, col=l15
                const int kk = (p >> 2) * 16 + hi * 8 + (p & 3) * 2;
                b[t].u[p] = *(const unsigned*)&lB[(wn + t * 16 + l15) * 32 + kk];
            }
#pragma unroll
        for (int t = 0; t < NT; ++t)
            c[t] = __builtin_amdgcn_wmma_f32_16x16x32_bf16(false, a.v, false, b[t].v,
                                                           (short)0, c[t], false, false);
        __syncthreads();
    }
}

// ---------------------------------------------------------------------------
// Converters
// ---------------------------------------------------------------------------
__global__ void __launch_bounds__(256) k_cvt_bf16(const float* __restrict__ in,
                                                  unsigned short* __restrict__ out)
{
    const size_t i = ((size_t)blockIdx.x * 256 + threadIdx.x) * 4;
    float4 f = *(const float4*)(in + i);
    out[i + 0] = f2bf(f.x); out[i + 1] = f2bf(f.y);
    out[i + 2] = f2bf(f.z); out[i + 3] = f2bf(f.w);
}

// Wt[d][e] = W[e][d], f32 -> bf16, tiled through LDS
__global__ void __launch_bounds__(256) k_transpose_w(const float* __restrict__ W,
                                                     unsigned short* __restrict__ Wt)
{
    __shared__ float t[32][33];
    const int bx = blockIdx.x * 32, by = blockIdx.y * 32;
    const int tx = threadIdx.x & 31, ty = threadIdx.x >> 5;  // 32 x 8
#pragma unroll
    for (int r = 0; r < 32; r += 8)
        t[ty + r][tx] = W[(size_t)(by + ty + r) * 1024 + bx + tx];
    __syncthreads();
#pragma unroll
    for (int r = 0; r < 32; r += 8)
        Wt[(size_t)(bx + ty + r) * 1024 + by + tx] = f2bf(t[tx][ty + r]);
}

// ---------------------------------------------------------------------------
// GEMM: C = A(bf16 [M,1024]) * B(bf16 [1024,1024]) + bias, block tile 64x128
// MODE 0: f32 row-major [M,1024]        (output projection -> d_out)
// MODE 1: bf16 scatter  [B,H,L,DH]      (Q, V)
// MODE 2: bf16 scatter  [B,H,DH,L]      (K transposed)
// ---------------------------------------------------------------------------
template<int MODE>
__global__ void __launch_bounds__(256) k_gemm_xw(
    const unsigned short* __restrict__ A,
    const unsigned short* __restrict__ Bw,
    const float* __restrict__ bias,
    void* __restrict__ out, int Kdim)
{
    const int rowBase = blockIdx.y * 64;
    const int colBase = blockIdx.x * 128;
    v8f c[4];
    wmma_tile<4, false>(A, 1024, Bw, 1024, Kdim, rowBase, colBase, c);

    const int tid = threadIdx.x, lane = tid & 31, wave = tid >> 5;
    const int wm = (wave >> 1) * 16, wn = (wave & 1) * 64;
    const int l15 = lane & 15, hi = lane >> 4;

#pragma unroll
    for (int t = 0; t < 4; ++t) {
        const int col = colBase + wn + t * 16 + l15;
        const float bv = bias[col];
#pragma unroll
        for (int r = 0; r < 8; ++r) {
            const int m = rowBase + wm + hi * 8 + r;
            const float v = c[t][r] + bv;
            if (MODE == 0) {
                ((float*)out)[(size_t)m * 1024 + col] = v;
            } else {
                unsigned short* o = (unsigned short*)out;
                const int b = m >> 10, l = m & 1023;
                if (MODE == 1)
                    o[(((size_t)b * 16 + (col >> 6)) * 1024 + l) * 64 + (col & 63)] = f2bf(v);
                else
                    o[(((size_t)b * 16 + (col >> 6)) * 64 + (col & 63)) * 1024 + l] = f2bf(v);
            }
        }
    }
}

// ---------------------------------------------------------------------------
// Scores: S[z] = (Q[z] * Kt[z]) / 8, z = b*16+h; Q [L,64], Kt [64,L]
// ---------------------------------------------------------------------------
__global__ void __launch_bounds__(256) k_scores(
    const unsigned short* __restrict__ Q,
    const unsigned short* __restrict__ Kt,
    float* __restrict__ S)
{
    const int z = blockIdx.z;
    const unsigned short* Az = Q + (size_t)z * 1024 * 64;
    const unsigned short* Bz = Kt + (size_t)z * 64 * 1024;
    float* Sz = S + (size_t)z * 1024 * 1024;
    const int rowBase = blockIdx.y * 64;
    const int colBase = blockIdx.x * 128;
    v8f c[4];
    wmma_tile<4, false>(Az, 64, Bz, 1024, 64, rowBase, colBase, c);

    const int tid = threadIdx.x, lane = tid & 31, wave = tid >> 5;
    const int wm = (wave >> 1) * 16, wn = (wave & 1) * 64;
    const int l15 = lane & 15, hi = lane >> 4;
#pragma unroll
    for (int t = 0; t < 4; ++t) {
        const int col = colBase + wn + t * 16 + l15;
#pragma unroll
        for (int r = 0; r < 8; ++r) {
            const int m = rowBase + wm + hi * 8 + r;
            Sz[(size_t)m * 1024 + col] = c[t][r] * 0.125f;
        }
    }
}

// ---------------------------------------------------------------------------
// Softmax with masks, in place over A. One 256-thread block per (z, q) row.
// Pre-softmax: absent keys excluded (-inf). Post-softmax: absent query -> 0.
// ---------------------------------------------------------------------------
__global__ void __launch_bounds__(256) k_softmax(float* __restrict__ Aattn,
                                                 const float* __restrict__ bemb)
{
    __shared__ float red[256];
    const int row = blockIdx.x;                  // row = z*1024 + q, z = b*16 + h
    const int b = row >> 14;
    const int q = row & 1023;
    float* rp = Aattn + (size_t)row * 1024;
    const float* bb = bemb + (size_t)b * 1024;
    const int t = threadIdx.x;

    if (!(bb[q] > 0.0f)) {                       // absent query row -> zeros
        float4 zz = {0.f, 0.f, 0.f, 0.f};
        ((float4*)rp)[t] = zz;
        return;
    }

    float4 v  = ((const float4*)rp)[t];
    float4 kb = ((const float4*)bb)[t];
    const bool p0 = kb.x > 0.f, p1 = kb.y > 0.f, p2 = kb.z > 0.f, p3 = kb.w > 0.f;

    float m = -__builtin_inff();
    if (p0) m = fmaxf(m, v.x);
    if (p1) m = fmaxf(m, v.y);
    if (p2) m = fmaxf(m, v.z);
    if (p3) m = fmaxf(m, v.w);

    red[t] = m; __syncthreads();
#pragma unroll
    for (int s = 128; s > 0; s >>= 1) {
        if (t < s) red[t] = fmaxf(red[t], red[t + s]);
        __syncthreads();
    }
    const float mx = red[0];
    __syncthreads();

    const float e0 = p0 ? __expf(v.x - mx) : 0.f;
    const float e1 = p1 ? __expf(v.y - mx) : 0.f;
    const float e2 = p2 ? __expf(v.z - mx) : 0.f;
    const float e3 = p3 ? __expf(v.w - mx) : 0.f;

    red[t] = e0 + e1 + e2 + e3; __syncthreads();
#pragma unroll
    for (int s = 128; s > 0; s >>= 1) {
        if (t < s) red[t] += red[t + s];
        __syncthreads();
    }
    const float inv = 1.0f / red[0];

    float4 o = {e0 * inv, e1 * inv, e2 * inv, e3 * inv};
    ((float4*)rp)[t] = o;
}

// ---------------------------------------------------------------------------
// ctx[b, l, h*64+dh] = sum_k A[z, l, k] * V[z, k, dh]   (A f32 -> bf16 on the fly)
// ---------------------------------------------------------------------------
__global__ void __launch_bounds__(256) k_av(
    const float* __restrict__ Aattn,
    const unsigned short* __restrict__ V,
    unsigned short* __restrict__ ctx)
{
    const int z = blockIdx.z;
    const int b = z >> 4, h = z & 15;
    const float* Az = Aattn + (size_t)z * 1024 * 1024;
    const unsigned short* Bz = V + (size_t)z * 1024 * 64;
    const int rowBase = blockIdx.y * 64;
    v8f c[2];
    wmma_tile<2, true>(Az, 1024, Bz, 64, 1024, rowBase, 0, c);

    const int tid = threadIdx.x, lane = tid & 31, wave = tid >> 5;
    const int wm = (wave >> 1) * 16, wn = (wave & 1) * 32;
    const int l15 = lane & 15, hi = lane >> 4;
#pragma unroll
    for (int t = 0; t < 2; ++t)
#pragma unroll
        for (int r = 0; r < 8; ++r) {
            const int l = rowBase + wm + hi * 8 + r;
            ctx[((size_t)b * 1024 + l) * 1024 + h * 64 + wn + t * 16 + l15] = f2bf(c[t][r]);
        }
}

// ---------------------------------------------------------------------------
// Launch
// ---------------------------------------------------------------------------
extern "C" void kernel_launch(void* const* d_in, const int* in_sizes, int n_in,
                              void* d_out, int out_size, void* d_ws, size_t ws_size,
                              hipStream_t stream) {
    const float* x    = (const float*)d_in[0];
    const float* bemb = (const float*)d_in[1];
    const float* Wq   = (const float*)d_in[2];
    const float* Wqb  = (const float*)d_in[3];
    const float* Wk   = (const float*)d_in[4];
    const float* Wkb  = (const float*)d_in[5];
    const float* Wv   = (const float*)d_in[6];
    const float* Wvb  = (const float*)d_in[7];
    const float* Wo   = (const float*)d_in[8];
    const float* Wob  = (const float*)d_in[9];

    const size_t MB = (size_t)1 << 20;
    char* ws = (char*)d_ws;
    unsigned short* xb  = (unsigned short*)(ws + 0 * MB);    // 8 MiB  bf16 x [4096,1024]
    unsigned short* Wtq = (unsigned short*)(ws + 8 * MB);    // 2 MiB each
    unsigned short* Wtk = (unsigned short*)(ws + 10 * MB);
    unsigned short* Wtv = (unsigned short*)(ws + 12 * MB);
    unsigned short* Wto = (unsigned short*)(ws + 14 * MB);
    unsigned short* Qb  = (unsigned short*)(ws + 16 * MB);   // 8 MiB [B,H,L,DH]
    unsigned short* Ktb = (unsigned short*)(ws + 24 * MB);   // 8 MiB [B,H,DH,L]
    unsigned short* Vb  = (unsigned short*)(ws + 32 * MB);   // 8 MiB [B,H,L,DH]
    unsigned short* ctx = (unsigned short*)(ws + 40 * MB);   // 8 MiB [B,L,D]

    float* outp = (float*)d_out;                              // [4096,1024] f32
    float* Aout = outp + (size_t)4 * 1024 * 1024;             // [64,1024,1024] f32

    const dim3 blk(256);

    k_cvt_bf16<<<4096, blk, 0, stream>>>(x, xb);
    k_transpose_w<<<dim3(32, 32), blk, 0, stream>>>(Wq, Wtq);
    k_transpose_w<<<dim3(32, 32), blk, 0, stream>>>(Wk, Wtk);
    k_transpose_w<<<dim3(32, 32), blk, 0, stream>>>(Wv, Wtv);
    k_transpose_w<<<dim3(32, 32), blk, 0, stream>>>(Wo, Wto);

    k_gemm_xw<1><<<dim3(8, 64), blk, 0, stream>>>(xb, Wtq, Wqb, (void*)Qb, 1024);
    k_gemm_xw<2><<<dim3(8, 64), blk, 0, stream>>>(xb, Wtk, Wkb, (void*)Ktb, 1024);
    k_gemm_xw<1><<<dim3(8, 64), blk, 0, stream>>>(xb, Wtv, Wvb, (void*)Vb, 1024);

    k_scores<<<dim3(8, 16, 64), blk, 0, stream>>>(Qb, Ktb, Aout);
    k_softmax<<<65536, blk, 0, stream>>>(Aout, bemb);
    k_av<<<dim3(1, 16, 64), blk, 0, stream>>>(Aout, Vb, ctx);

    k_gemm_xw<0><<<dim3(8, 64), blk, 0, stream>>>(ctx, Wto, Wob, d_out, 1024);
}
